// SDPAttentionPairBias_37546604101753
// MI455X (gfx1250) — matrix operand order repaired and checked by reading, not compiled, not run on hardware
//
#include <hip/hip_runtime.h>

#define B_  2
#define N_  1024
#define CS_ 512
#define CZ_ 128
#define H_  16
#define D_  32

typedef __attribute__((ext_vector_type(16))) __bf16 v16bf;
typedef __attribute__((ext_vector_type(8)))  float  v8f;

union ABf { v16bf v; unsigned int u[8]; unsigned short s[16]; };
union V8  { v8f v; float f[8]; };

// hardware bf16 converts (v_cvt_pk_bf16_f32 / v_cvt_bf16_f32) instead of
// manual RNE bit-twiddling
__device__ __forceinline__ unsigned short f2bf(float f) {
    union { __bf16 h; unsigned short u; } t;
    t.h = (__bf16)f;
    return t.u;
}
__device__ __forceinline__ unsigned int packbf(float a, float b) {
    union { __bf16 h[2]; unsigned int u; } t;
    t.h[0] = (__bf16)a;
    t.h[1] = (__bf16)b;
    return t.u;
}
__device__ __forceinline__ v8f wmma_bf(const ABf& a, const ABf& b, v8f c) {
    return __builtin_amdgcn_wmma_f32_16x16x32_bf16(false, a.v, false, b.v,
                                                   (short)0, c, false, false);
}
// async global->LDS copy (ASYNCcnt-tracked, no VGPR round trip)
__device__ __forceinline__ void async_lds_b128(void* lds, const void* gptr) {
    unsigned ldsoff = (unsigned)(unsigned long long)lds;
    unsigned long long ga = (unsigned long long)gptr;
    asm volatile("global_load_async_to_lds_b128 %0, %1, off"
                 :: "v"(ldsoff), "v"(ga) : "memory");
}
__device__ __forceinline__ void wait_asynccnt0() {
    asm volatile("s_wait_asynccnt 0" ::: "memory");
}

// ---------------------------------------------------------------- prep ----
__global__ __launch_bounds__(256) void cvt_bf16_kernel(
    const float* __restrict__ src, unsigned short* __restrict__ dst, int n) {
    int i = (blockIdx.x * 256 + threadIdx.x) * 4;
    if (i + 3 < n) {
        float4 f = *(const float4*)(src + i);
        *(unsigned int*)(dst + i)     = packbf(f.x, f.y);
        *(unsigned int*)(dst + i + 2) = packbf(f.z, f.w);
    }
}

__global__ __launch_bounds__(256) void prep_wz_kernel(
    const float* __restrict__ Wz, const float* __restrict__ gamma,
    const float* __restrict__ beta, unsigned short* __restrict__ gw,
    float* __restrict__ GhBh) {
    int t = threadIdx.x;
    for (int idx = t; idx < H_ * CZ_; idx += 256) {
        int c = idx & (CZ_ - 1);
        gw[idx] = f2bf(Wz[idx] * gamma[c]);
    }
    if (t < 32) {
        int hh = t & 15;
        const float* w = Wz + hh * CZ_;
        float acc = 0.f;
        if (t < 16) { for (int c = 0; c < CZ_; ++c) acc += gamma[c] * w[c]; }
        else        { for (int c = 0; c < CZ_; ++c) acc += beta[c]  * w[c]; }
        GhBh[t] = acc;
    }
}

// ------------------------------------------------------ projection GEMM ----
// out[r][c] = sum_k X[r][k] * W[c][k]   (rows = B*N = 2048, cols = K = 512)
// mode 0: +bq, bf16 out (B,H,N,D)      mode 1: bf16 out (B,H,N,D)
// mode 2: bf16 out (B,H,D,N)           mode 3: sigmoid, f32 out (B,N,CS)
// mode 4: f32 out (B,N,CS), bf16 input
__global__ __launch_bounds__(256) void proj_gemm(
    const float* __restrict__ X, const unsigned short* __restrict__ Xbf,
    const unsigned short* __restrict__ W, const float* __restrict__ bq,
    float* __restrict__ outF, unsigned short* __restrict__ outBf, int mode) {
    __shared__ unsigned short As[128][36];
    __shared__ unsigned short Bs[64][36];

    const int t = threadIdx.x;
    const int lane = t & 31, wave = t >> 5;
    const int wr = wave >> 1, wc = wave & 1;
    const int r0 = (blockIdx.x >> 3) * 128;
    const int c0 = (blockIdx.x & 7) * 64;

    const int M       = lane & 15;
    const int abase   = (lane < 16) ? 0 : 8;
    const int kb      = (lane < 16) ? 0 : 16;
    const int halfAdd = (lane < 16) ? 0 : 8;

    V8 acc[2][2];
    for (int a = 0; a < 2; ++a)
        for (int bb2 = 0; bb2 < 2; ++bb2)
            for (int r = 0; r < 8; ++r) acc[a][bb2].f[r] = 0.f;

    const int arow = t >> 1, akseg = (t & 1) * 16;
    const int brow = t >> 2, bkseg = (t & 3) * 8;

    for (int kk = 0; kk < CS_; kk += 32) {
        __syncthreads();
        // B tile (weights, already bf16): async copy straight to LDS
        async_lds_b128(&Bs[brow][bkseg],
                       W + (size_t)(c0 + brow) * CS_ + kk + bkseg);
        if (X) {  // fp32 input path: load, convert with v_cvt_pk_bf16, stage
            const float* p = X + (size_t)(r0 + arow) * CS_ + kk + akseg;
            if (kk + 32 < CS_) __builtin_prefetch(p + 32, 0, 1);
            float4 f0 = ((const float4*)p)[0];
            float4 f1 = ((const float4*)p)[1];
            float4 f2 = ((const float4*)p)[2];
            float4 f3 = ((const float4*)p)[3];
            unsigned int* ar = (unsigned int*)&As[arow][akseg];
            ar[0] = packbf(f0.x, f0.y); ar[1] = packbf(f0.z, f0.w);
            ar[2] = packbf(f1.x, f1.y); ar[3] = packbf(f1.z, f1.w);
            ar[4] = packbf(f2.x, f2.y); ar[5] = packbf(f2.z, f2.w);
            ar[6] = packbf(f3.x, f3.y); ar[7] = packbf(f3.z, f3.w);
        } else {  // bf16 input path: async copy both halves to LDS
            const unsigned short* p = Xbf + (size_t)(r0 + arow) * CS_ + kk + akseg;
            async_lds_b128(&As[arow][akseg], p);
            async_lds_b128(&As[arow][akseg + 8], p + 8);
        }
        wait_asynccnt0();
        __syncthreads();

        ABf a[2], bmat[2];
#pragma unroll
        for (int mt = 0; mt < 2; ++mt) {
            const unsigned short* ap = &As[wr * 32 + mt * 16 + M][0];
#pragma unroll
            for (int h2 = 0; h2 < 2; ++h2)
#pragma unroll
                for (int pp = 0; pp < 4; ++pp)
                    a[mt].u[h2 * 4 + pp] =
                        *(const unsigned int*)&ap[h2 * 16 + abase + 2 * pp];
        }
#pragma unroll
        for (int nt = 0; nt < 2; ++nt) {
            const unsigned short* bp = &Bs[wc * 32 + nt * 16 + M][0];
#pragma unroll
            for (int v = 0; v < 8; ++v)
                bmat[nt].u[v] = *(const unsigned int*)&bp[(v >> 2) * 16 + abase + 2 * (v & 3)];
        }
#pragma unroll
        for (int mt = 0; mt < 2; ++mt)
#pragma unroll
            for (int nt = 0; nt < 2; ++nt)
                acc[mt][nt].v = wmma_bf(a[mt], bmat[nt], acc[mt][nt].v);
    }

#pragma unroll
    for (int mt = 0; mt < 2; ++mt)
#pragma unroll
        for (int nt = 0; nt < 2; ++nt) {
            int cg = c0 + wc * 32 + nt * 16 + M;
#pragma unroll
            for (int rr = 0; rr < 8; ++rr) {
                int rg = r0 + wr * 32 + mt * 16 + rr + halfAdd;
                float val = acc[mt][nt].f[rr];
                int b = rg >> 10, n = rg & (N_ - 1);
                int hh = cg >> 5, d = cg & (D_ - 1);
                if (mode == 0) {
                    val += bq[cg];
                    outBf[(((size_t)b * H_ + hh) * N_ + n) * D_ + d] = f2bf(val);
                } else if (mode == 1) {
                    outBf[(((size_t)b * H_ + hh) * N_ + n) * D_ + d] = f2bf(val);
                } else if (mode == 2) {
                    outBf[(((size_t)b * H_ + hh) * D_ + d) * N_ + n] = f2bf(val);
                } else if (mode == 3) {
                    outF[(size_t)rg * CS_ + cg] = 1.0f / (1.0f + __expf(-val));
                } else {
                    outF[(size_t)rg * CS_ + cg] = val;
                }
            }
        }
}

// --------------------------------------------- pair bias: LN(z) @ (g*Wz)T --
__global__ __launch_bounds__(256) void pair_bias_kernel(
    const float* __restrict__ z, const unsigned short* __restrict__ gw,
    const float* __restrict__ GhBh, _Float16* __restrict__ biasOut) {
    __shared__ float stats[8][2][16];
    const int t = threadIdx.x, lane = t & 31, wave = t >> 5;
    const size_t row0 = ((size_t)blockIdx.x * 8 + wave) * 16;
    const int b  = (int)(row0 >> 20);
    const int i  = (int)((row0 >> 10) & (N_ - 1));
    const int j0 = (int)(row0 & (N_ - 1));
    const int M       = lane & 15;
    const int kb      = (lane < 16) ? 0 : 16;
    const int abase   = (lane < 16) ? 0 : 8;
    const int halfAdd = (lane < 16) ? 0 : 8;

    ABf bw[4];
#pragma unroll
    for (int c4 = 0; c4 < 4; ++c4)
#pragma unroll
        for (int v = 0; v < 8; ++v)
            bw[c4].u[v] = *(const unsigned int*)&gw[M * CZ_ + c4 * 32 + (v >> 2) * 16 + abase + 2 * (v & 3)];

    V8 acc;
    for (int r = 0; r < 8; ++r) acc.f[r] = 0.f;
    float sum = 0.f, sumsq = 0.f;
    const float* zrow = z + (row0 + M) * CZ_;
#pragma unroll
    for (int c4 = 0; c4 < 4; ++c4) {
        ABf a;
        const float* zp = zrow + c4 * 32 + abase;
#pragma unroll
        for (int h2 = 0; h2 < 2; ++h2) {
            float4 f0 = ((const float4*)(zp + h2 * 16))[0];
            float4 f1 = ((const float4*)(zp + h2 * 16))[1];
            sum   += f0.x + f0.y + f0.z + f0.w + f1.x + f1.y + f1.z + f1.w;
            sumsq += f0.x*f0.x + f0.y*f0.y + f0.z*f0.z + f0.w*f0.w
                   + f1.x*f1.x + f1.y*f1.y + f1.z*f1.z + f1.w*f1.w;
            a.u[h2 * 4 + 0] = packbf(f0.x, f0.y);
            a.u[h2 * 4 + 1] = packbf(f0.z, f0.w);
            a.u[h2 * 4 + 2] = packbf(f1.x, f1.y);
            a.u[h2 * 4 + 3] = packbf(f1.z, f1.w);
        }
        acc.v = wmma_bf(a, bw[c4], acc.v);
    }
    sum   += __shfl_xor(sum, 16);
    sumsq += __shfl_xor(sumsq, 16);
    if (lane < 16) {
        float mean = sum * (1.0f / CZ_);
        float var  = sumsq * (1.0f / CZ_) - mean * mean;
        stats[wave][0][lane] = mean;
        stats[wave][1][lane] = rsqrtf(var + 1e-5f);
    }
    __syncthreads();
    const float Gh = GhBh[M], Bh = GhBh[16 + M];
#pragma unroll
    for (int rr = 0; rr < 8; ++rr) {
        int Mr = rr + halfAdd;
        float mean = stats[wave][0][Mr], rs = stats[wave][1][Mr];
        float val = (acc.f[rr] - mean * Gh) * rs + Bh;
        int j = j0 + Mr;
        biasOut[(((size_t)b * N_ + i) * N_ + j) * H_ + M] = (_Float16)val;
    }
}

// ----------------------------------------------------- flash attention ----
__global__ __launch_bounds__(512) void attn_kernel(
    const unsigned short* __restrict__ Qb, const unsigned short* __restrict__ Kb,
    const unsigned short* __restrict__ Vt, const _Float16* __restrict__ bias,
    const float* __restrict__ mask, const float* __restrict__ G,
    unsigned short* __restrict__ og) {
    __shared__ unsigned short Pt[16][16][40];  // [head][row][32 cols + pad]
    const int t = threadIdx.x, lane = t & 31, h = t >> 5;
    const int b  = blockIdx.x >> 6;
    const int i0 = (blockIdx.x & 63) * 16;
    const int M       = lane & 15;
    const int abase   = (lane < 16) ? 0 : 8;
    const int kb16    = (lane < 16) ? 0 : 16;
    const int halfAdd = (lane < 16) ? 0 : 8;
    const size_t headOff = ((size_t)b * H_ + h) * N_;

    ABf qa;
    {
        const unsigned short* qp = Qb + (headOff + i0 + M) * D_ + abase;
        uint4 q0 = *(const uint4*)qp;
        uint4 q1 = *(const uint4*)(qp + 16);
        qa.u[0] = q0.x; qa.u[1] = q0.y; qa.u[2] = q0.z; qa.u[3] = q0.w;
        qa.u[4] = q1.x; qa.u[5] = q1.y; qa.u[6] = q1.z; qa.u[7] = q1.w;
    }

    float mrow[8], lrow[8];
    V8 acc0, acc1;
    for (int r = 0; r < 8; ++r) { mrow[r] = -1e30f; lrow[r] = 0.f; acc0.f[r] = 0.f; acc1.f[r] = 0.f; }
    const float scale = 0.17677669529663689f;  // 1/sqrt(32)

    for (int jt = 0; jt < N_ / 32; ++jt) {
        V8 S[2];
#pragma unroll
        for (int sub = 0; sub < 2; ++sub) {
            const int jc = jt * 32 + sub * 16;
            ABf kf;
            const unsigned short* kp = Kb + (headOff + jc + M) * D_ + abase;
            uint4 k0 = *(const uint4*)kp;
            uint4 k1 = *(const uint4*)(kp + 16);
            kf.u[0] = k0.x; kf.u[1] = k0.y; kf.u[2] = k0.z; kf.u[3] = k0.w;
            kf.u[4] = k1.x; kf.u[5] = k1.y; kf.u[6] = k1.z; kf.u[7] = k1.w;
            v8f zero = {0.f, 0.f, 0.f, 0.f, 0.f, 0.f, 0.f, 0.f};
            V8 s; s.v = wmma_bf(qa, kf, zero);
            float mk   = mask[b * N_ + jc + M];
            float madd = (1.0f - mk) * (-1000000.0f);
            const _Float16* bp =
                bias + (((size_t)b * N_ + i0) * N_ + jc + M) * H_ + h;
#pragma unroll
            for (int rr = 0; rr < 8; ++rr) {
                int irow = rr + halfAdd;
                float bv = (float)bp[(size_t)irow * (N_ * H_)];
                s.f[rr] = s.f[rr] * scale + bv + madd;
            }
            S[sub] = s;
        }
#pragma unroll
        for (int rr = 0; rr < 8; ++rr) {
            float mx = fmaxf(S[0].f[rr], S[1].f[rr]);
            mx = fmaxf(mx, __shfl_xor(mx, 1));
            mx = fmaxf(mx, __shfl_xor(mx, 2));
            mx = fmaxf(mx, __shfl_xor(mx, 4));
            mx = fmaxf(mx, __shfl_xor(mx, 8));
            float mn = fmaxf(mrow[rr], mx);
            float cf = __expf(mrow[rr] - mn);
            mrow[rr] = mn;
            float p0 = __expf(S[0].f[rr] - mn);
            float p1 = __expf(S[1].f[rr] - mn);
            S[0].f[rr] = p0; S[1].f[rr] = p1;
            float rs = p0 + p1;
            rs += __shfl_xor(rs, 1);
            rs += __shfl_xor(rs, 2);
            rs += __shfl_xor(rs, 4);
            rs += __shfl_xor(rs, 8);
            lrow[rr] = lrow[rr] * cf + rs;
            acc0.f[rr] *= cf; acc1.f[rr] *= cf;
        }
        // transpose P (C layout -> A layout) through LDS
#pragma unroll
        for (int rr = 0; rr < 8; ++rr) {
            int Mr = rr + halfAdd;
            Pt[h][Mr][M]      = f2bf(S[0].f[rr]);
            Pt[h][Mr][16 + M] = f2bf(S[1].f[rr]);
        }
        asm volatile("s_wait_dscnt 0" ::: "memory");
        ABf pa;
        {
            uint4 p0 = *(const uint4*)&Pt[h][M][abase];
            uint4 p1 = *(const uint4*)&Pt[h][M][16 + abase];
            pa.u[0] = p0.x; pa.u[1] = p0.y; pa.u[2] = p0.z; pa.u[3] = p0.w;
            pa.u[4] = p1.x; pa.u[5] = p1.y; pa.u[6] = p1.z; pa.u[7] = p1.w;
        }
#pragma unroll
        for (int fd = 0; fd < 2; ++fd) {
            ABf vf;
            const unsigned short* vp =
                Vt + (((size_t)b * H_ + h) * D_ + fd * 16 + M) * N_ + jt * 32 + abase;
            uint4 v0 = *(const uint4*)vp;
            uint4 v1 = *(const uint4*)(vp + 16);
            vf.u[0] = v0.x; vf.u[1] = v0.y; vf.u[2] = v0.z; vf.u[3] = v0.w;
            vf.u[4] = v1.x; vf.u[5] = v1.y; vf.u[6] = v1.z; vf.u[7] = v1.w;
            if (fd == 0) acc0.v = wmma_bf(pa, vf, acc0.v);
            else         acc1.v = wmma_bf(pa, vf, acc1.v);
        }
    }
#pragma unroll
    for (int rr = 0; rr < 8; ++rr) {
        int irow = i0 + rr + halfAdd;
        float inv = 1.0f / lrow[rr];
        size_t base = ((size_t)b * N_ + irow) * CS_ + h * D_;
        og[base + M]      = f2bf(acc0.f[rr] * inv * G[base + M]);
        og[base + 16 + M] = f2bf(acc1.f[rr] * inv * G[base + 16 + M]);
    }
}

// -------------------------------------------------------------- launch ----
extern "C" void kernel_launch(void* const* d_in, const int* in_sizes, int n_in,
                              void* d_out, int out_size, void* d_ws, size_t ws_size,
                              hipStream_t stream) {
    const float* s     = (const float*)d_in[0];
    const float* z     = (const float*)d_in[1];
    const float* mask  = (const float*)d_in[2];
    const float* k_in  = (const float*)d_in[3];
    const float* Wq    = (const float*)d_in[4];
    const float* bq    = (const float*)d_in[5];
    const float* Wk    = (const float*)d_in[6];
    const float* Wv    = (const float*)d_in[7];
    const float* Wg    = (const float*)d_in[8];
    const float* gamma = (const float*)d_in[9];
    const float* beta  = (const float*)d_in[10];
    const float* Wz    = (const float*)d_in[11];
    const float* Wo    = (const float*)d_in[12];

    char* ws = (char*)d_ws;
    size_t off = 0;
    auto alloc = [&](size_t bytes) {
        char* p = ws + off;
        off = (off + bytes + 255) & ~(size_t)255;
        return p;
    };
    unsigned short* Qb   = (unsigned short*)alloc((size_t)B_ * H_ * N_ * D_ * 2);
    unsigned short* Kbuf = (unsigned short*)alloc((size_t)B_ * H_ * N_ * D_ * 2);
    unsigned short* Vt   = (unsigned short*)alloc((size_t)B_ * H_ * N_ * D_ * 2);
    float*          Gbuf = (float*)alloc((size_t)B_ * N_ * CS_ * 4);
    unsigned short* Wqb  = (unsigned short*)alloc((size_t)CS_ * CS_ * 2);
    unsigned short* Wkb  = (unsigned short*)alloc((size_t)CS_ * CS_ * 2);
    unsigned short* Wvb  = (unsigned short*)alloc((size_t)CS_ * CS_ * 2);
    unsigned short* Wgb  = (unsigned short*)alloc((size_t)CS_ * CS_ * 2);
    unsigned short* Wob  = (unsigned short*)alloc((size_t)CS_ * CS_ * 2);
    unsigned short* gw   = (unsigned short*)alloc((size_t)H_ * CZ_ * 2);
    float*          GhBh = (float*)alloc(32 * 4);
    _Float16*       biasB = (_Float16*)alloc((size_t)B_ * N_ * N_ * H_ * 2);
    unsigned short* og   = (unsigned short*)alloc((size_t)B_ * N_ * CS_ * 2);

    const int nW = CS_ * CS_;
    cvt_bf16_kernel<<<nW / 1024, 256, 0, stream>>>(Wq, Wqb, nW);
    cvt_bf16_kernel<<<nW / 1024, 256, 0, stream>>>(Wk, Wkb, nW);
    cvt_bf16_kernel<<<nW / 1024, 256, 0, stream>>>(Wv, Wvb, nW);
    cvt_bf16_kernel<<<nW / 1024, 256, 0, stream>>>(Wg, Wgb, nW);
    cvt_bf16_kernel<<<nW / 1024, 256, 0, stream>>>(Wo, Wob, nW);
    prep_wz_kernel<<<1, 256, 0, stream>>>(Wz, gamma, beta, gw, GhBh);

    proj_gemm<<<128, 256, 0, stream>>>(s,    nullptr, Wqb, bq,      nullptr, Qb,   0);
    proj_gemm<<<128, 256, 0, stream>>>(k_in, nullptr, Wkb, nullptr, nullptr, Kbuf, 1);
    proj_gemm<<<128, 256, 0, stream>>>(k_in, nullptr, Wvb, nullptr, nullptr, Vt,   2);
    proj_gemm<<<128, 256, 0, stream>>>(s,    nullptr, Wgb, nullptr, Gbuf,    nullptr, 3);

    pair_bias_kernel<<<(B_ * N_ * N_) / (8 * 16), 256, 0, stream>>>(z, gw, GhBh, biasB);

    attn_kernel<<<B_ * (N_ / 16), 512, 0, stream>>>(Qb, Kbuf, Vt, biasB, mask,
                                                    Gbuf, og);

    proj_gemm<<<128, 256, 0, stream>>>(nullptr, og, Wob, nullptr, (float*)d_out,
                                       nullptr, 4);
}
